// BiDAF_25615184953429
// MI455X (gfx1250) — compile-verified, matrix operand order checked by reading, not run on hardware
//
#include <hip/hip_runtime.h>
#include <hip/hip_bf16.h>
#include <math.h>

typedef __attribute__((ext_vector_type(16))) _Float16     v16h;
typedef __attribute__((ext_vector_type(8)))  float        v8f;
typedef __attribute__((ext_vector_type(8)))  unsigned int v8u;

__device__ __forceinline__ float sigm(float x) { return 1.f / (1.f + __expf(-x)); }
__device__ __forceinline__ float ftanh(float x) {
    x = fminf(fmaxf(x, -15.f), 15.f);
    float e = __expf(2.f * x);
    return (e - 1.f) / (e + 1.f);
}
__device__ __forceinline__ unsigned pack2(float a, float b) {
    union { _Float16 h[2]; unsigned u; } x;
    x.h[0] = (_Float16)a; x.h[1] = (_Float16)b;
    return x.u;
}

// Build a 16x32 f16 fragment (A or B^T) from an LDS row holding 32 halves,
// using packed 32-bit reads. Half h of the fragment holds k =
// (h&7) + (h>=8?16:0) + (lane>=16?8:0)  ->  uint i holds k/2 = (i&3)+((i&4)?8:0)+lanehi*4.
__device__ __forceinline__ v16h frag_from_lds(const _Float16* rowbase, int lane) {
    const unsigned* pu = (const unsigned*)rowbase;
    const int hi = (lane & 16) ? 4 : 0;
    v8u u;
#pragma unroll
    for (int i = 0; i < 8; ++i)
        u[i] = pu[((i & 4) ? 8 : 0) + (i & 3) + hi];
    return __builtin_bit_cast(v16h, u);
}

// ---------------------------------------------------------------------------
// LDS-staged WMMA GEMM: C[M,N] = act(A[M,K] @ op(B) + bias1 + bias2
//                                    + rowAdd[m] + colAdd[n]),  A scaled by ascale[k].
// transB=1: op(B)[k][n] = B[n*K+k] ; transB=0: op(B)[k][n] = B[k*N+n].
// Block 256 thr (8 waves), 64x64 block tile, wave = two 16x16 tiles.
// M must be a multiple of 64 (true for all uses). N,K arbitrary.
// ---------------------------------------------------------------------------
template <int K, int TRANSB, int ACT>
__global__ __launch_bounds__(256) void wmma_gemm(
    const float* __restrict__ A, const float* __restrict__ B, float* __restrict__ C,
    int M, int N, long sA, long sB, long sC,
    const float* __restrict__ bias1, const float* __restrict__ bias2,
    const float* __restrict__ ascale,
    const float* __restrict__ rowAdd, long sRow,
    const float* __restrict__ colAdd, long sCol)
{
    constexpr int LDK = 40;                  // padded half-stride (bank-conflict-free)
    __shared__ _Float16 As[64 * LDK];
    __shared__ _Float16 Bs[64 * LDK];

    const int  tid  = threadIdx.x;
    const int  lane = tid & 31, wave = tid >> 5;
    const long bb   = blockIdx.z;
    A += bb * sA; B += bb * sB; C += bb * sC;
    const float* rA = rowAdd ? rowAdd + bb * sRow : nullptr;
    const float* cA = colAdd ? colAdd + bb * sCol : nullptr;

    const int m0 = blockIdx.y * 64;
    const int n0 = blockIdx.x * 64;

    const int mt  = wave >> 1;               // shared m-tile for this wave's 2 tiles
    const int nt0 = (2 * wave) & 3, nt1 = nt0 + 1;

    const int ra = tid >> 2, ka = (tid & 3) * 8;  // staging map (A, B-trans): 8 contig k
    const int kb = tid >> 3, nb = (tid & 7) * 8;  // staging map (B no-trans): 8 contig n

    unsigned* Asu = (unsigned*)As;
    unsigned* Bsu = (unsigned*)Bs;
    const int sAu = ra * (LDK / 2) + ka / 2;      // uint index of this thread's chunk

    v8f acc0 = {}, acc1 = {};
    constexpr int KT = K & ~31;

    for (int k0 = 0; k0 < KT; k0 += 32) {
        {   // stage A (guard-free: m0+ra < M, kk < K in main loop)
            const float* ap = A + (long)(m0 + ra) * K + k0 + ka;
            float v[8];
#pragma unroll
            for (int j = 0; j < 8; ++j) {
                v[j] = ap[j];
                if (ascale) v[j] *= ascale[k0 + ka + j];
            }
#pragma unroll
            for (int j = 0; j < 4; ++j)
                Asu[sAu + j] = pack2(v[2 * j], v[2 * j + 1]);
        }
        if (TRANSB) {    // stage B as [n][k]; clamp+zero for n edge (branch-free)
            int n = n0 + ra;
            int nc = n < N ? n : N - 1;
            float sel = n < N ? 1.f : 0.f;
            const float* bp = B + (long)nc * K + k0 + ka;
#pragma unroll
            for (int j = 0; j < 4; ++j)
                Bsu[sAu + j] = pack2(bp[2 * j] * sel, bp[2 * j + 1] * sel);
        } else {
            const float* bp = B + (long)(k0 + kb) * N;
#pragma unroll
            for (int j = 0; j < 8; ++j) {
                int n = n0 + nb + j;
                int nc = n < N ? n : N - 1;
                float sel = n < N ? 1.f : 0.f;
                Bs[(nb + j) * LDK + kb] = (_Float16)(bp[nc] * sel);
            }
        }
        __syncthreads();
        v16h af = frag_from_lds(As + (mt * 16 + (lane & 15)) * LDK, lane);
        v16h b0 = frag_from_lds(Bs + (nt0 * 16 + (lane & 15)) * LDK, lane);
        v16h b1 = frag_from_lds(Bs + (nt1 * 16 + (lane & 15)) * LDK, lane);
        acc0 = __builtin_amdgcn_wmma_f32_16x16x32_f16(false, af, false, b0, (short)0, acc0, false, false);
        acc1 = __builtin_amdgcn_wmma_f32_16x16x32_f16(false, af, false, b1, (short)0, acc1, false, false);
        __syncthreads();
    }
    if constexpr ((K % 32) != 0) {           // constexpr tail (clamp+zero, no branches)
        const int k0 = KT;
#pragma unroll
        for (int j = 0; j < 8; ++j) {
            int kk = k0 + ka + j;
            int kc = kk < K ? kk : K - 1;
            float sel = kk < K ? 1.f : 0.f;
            float v = A[(long)(m0 + ra) * K + kc] * sel;
            if (ascale) v *= ascale[kc];
            As[ra * LDK + ka + j] = (_Float16)v;
        }
        if (TRANSB) {
            int n = n0 + ra;
            int nc = n < N ? n : N - 1;
#pragma unroll
            for (int j = 0; j < 8; ++j) {
                int kk = k0 + ka + j;
                int kc = kk < K ? kk : K - 1;
                float sel = (kk < K && n < N) ? 1.f : 0.f;
                Bs[ra * LDK + ka + j] = (_Float16)(B[(long)nc * K + kc] * sel);
            }
        } else {
            int kk = k0 + kb;
            int kc = kk < K ? kk : K - 1;
#pragma unroll
            for (int j = 0; j < 8; ++j) {
                int n = n0 + nb + j;
                int nc = n < N ? n : N - 1;
                float sel = (kk < K && n < N) ? 1.f : 0.f;
                Bs[(nb + j) * LDK + kb] = (_Float16)(B[(long)kc * N + nc] * sel);
            }
        }
        __syncthreads();
        v16h af = frag_from_lds(As + (mt * 16 + (lane & 15)) * LDK, lane);
        v16h b0 = frag_from_lds(Bs + (nt0 * 16 + (lane & 15)) * LDK, lane);
        v16h b1 = frag_from_lds(Bs + (nt1 * 16 + (lane & 15)) * LDK, lane);
        acc0 = __builtin_amdgcn_wmma_f32_16x16x32_f16(false, af, false, b0, (short)0, acc0, false, false);
        acc1 = __builtin_amdgcn_wmma_f32_16x16x32_f16(false, af, false, b1, (short)0, acc1, false, false);
    }
    // epilogue
#pragma unroll
    for (int half = 0; half < 2; ++half) {
        v8f acc = half ? acc1 : acc0;
        int n = n0 + (half ? nt1 : nt0) * 16 + (lane & 15);
        int nc = n < N ? n : N - 1;
        float badd = 0.f;
        if (bias1) badd += bias1[nc];
        if (bias2) badd += bias2[nc];
        if (cA)    badd += cA[nc];
#pragma unroll
        for (int r = 0; r < 8; ++r) {
            int m = m0 + mt * 16 + r + ((lane & 16) ? 8 : 0);
            float v = acc[r] + badd;
            if (rA) v += rA[m];
            if constexpr (ACT == 1) v = fmaxf(v, 0.f);
            if constexpr (ACT == 2) v = sigm(v);
            if (n < N) C[(long)m * N + n] = v;
        }
    }
}

// Pre-convert Whh [400,100] f32 -> [400,128] f16, zero-padded (guard-free scan reads).
__global__ void pad_weights_f16(const float* __restrict__ W, _Float16* __restrict__ out)
{
    int i = blockIdx.x * 256 + threadIdx.x;
    if (i < 400 * 128) {
        int r = i >> 7, c = i & 127;
        out[i] = (c < 100) ? (_Float16)W[r * 100 + c] : (_Float16)0.f;
    }
}

// ---------------------------------------------------------------------------
// Persistent BiLSTM scan. grid=2 (dir), block=512 (16 waves).
// z = XP[:,t,:] + h @ Whh^T via WMMA (M=32,N=400,K=100->padded 128).
// h in LDS as zero-padded f16; Whh fragments REGISTER-resident across all steps.
// XP is added in the gate phase (coalesced, immediate-offset loads), so the
// WMMA epilogue is pure ds_store. c in per-thread registers (fixed ownership).
// ---------------------------------------------------------------------------
__global__ __launch_bounds__(512) void lstm_scan(
    const float* __restrict__ XPf, const float* __restrict__ XPb,
    const _Float16* __restrict__ WhhPf, const _Float16* __restrict__ WhhPb,
    float* __restrict__ out, int T)
{
    __shared__ _Float16 hbuf[32 * 128];
    __shared__ float    zbuf[32 * 400];

    const int dir = blockIdx.x;
    const float*     XP  = dir ? XPb : XPf;
    const _Float16*  Whh = dir ? WhhPb : WhhPf;   // padded [400][128] f16
    const int outOff = dir ? 100 : 0;

    const int tid = threadIdx.x;
    const int lane = tid & 31, wave = tid >> 5;
    const int hi = (lane & 16) ? 4 : 0;

    for (int i = tid; i < 32 * 128; i += 512) hbuf[i] = (_Float16)0.f;
    float creg[7];
#pragma unroll
    for (int s = 0; s < 7; ++s) creg[s] = 0.f;
    // per-thread gate-cell ownership, hoisted out of the time loop
    int cb[7], cj[7];
#pragma unroll
    for (int s = 0; s < 7; ++s) {
        int cell = tid + s * 512;
        cb[s] = cell / 100; cj[s] = cell % 100;
    }

    // Preload this wave's Whh B-fragments into registers (<=4 tiles x 4 k-slabs).
    v8u bw[4][4];
#pragma unroll
    for (int ti = 0; ti < 4; ++ti) {
        int tile = wave + ti * 16;
        if (tile < 50) {
            int nt = tile >> 1;
            int ncol = nt * 16 + (lane & 15);
            const unsigned* wp = (const unsigned*)(Whh + (long)ncol * 128);
#pragma unroll
            for (int ks = 0; ks < 4; ++ks)
#pragma unroll
                for (int i = 0; i < 8; ++i)
                    bw[ti][ks][i] = wp[ks * 16 + ((i & 4) ? 8 : 0) + (i & 3) + hi];
        }
    }
    __syncthreads();

    for (int t = 0; t < T; ++t) {
        const int idx = dir ? (T - 1 - t) : t;
        // 2 m-tiles x 25 n-tiles = 50 tiles over 16 waves
#pragma unroll
        for (int ti = 0; ti < 4; ++ti) {
            int tile = wave + ti * 16;
            if (tile < 50) {
                int mt = tile & 1, nt = tile >> 1;
                int mrow = mt * 16 + (lane & 15);
                int ncol = nt * 16 + (lane & 15);
                const unsigned* hp = (const unsigned*)(hbuf + mrow * 128);
                v8f acc = {};
#pragma unroll
                for (int ks = 0; ks < 4; ++ks) {
                    v8u au;
#pragma unroll
                    for (int i = 0; i < 8; ++i)
                        au[i] = hp[ks * 16 + ((i & 4) ? 8 : 0) + (i & 3) + hi];
                    v16h a = __builtin_bit_cast(v16h, au);
                    v16h b = __builtin_bit_cast(v16h, bw[ti][ks]);
                    acc = __builtin_amdgcn_wmma_f32_16x16x32_f16(false, a, false, b, (short)0,
                                                                 acc, false, false);
                }
#pragma unroll
                for (int r = 0; r < 8; ++r) {
                    int m = mt * 16 + r + ((lane & 16) ? 8 : 0);
                    zbuf[m * 400 + ncol] = acc[r];
                }
            }
        }
        __syncthreads();
#pragma unroll
        for (int s = 0; s < 7; ++s) {
            if (tid + s * 512 < 3200) {
                int b = cb[s], j = cj[s];
                const float* xp = XP + ((long)b * T + idx) * 400 + j;
                float zi = zbuf[b * 400 + j]       + xp[0];
                float zf = zbuf[b * 400 + 100 + j] + xp[100];
                float zg = zbuf[b * 400 + 200 + j] + xp[200];
                float zo = zbuf[b * 400 + 300 + j] + xp[300];
                float c = sigm(zf) * creg[s] + sigm(zi) * ftanh(zg);
                float h = sigm(zo) * ftanh(c);
                creg[s] = c;
                hbuf[b * 128 + j] = (_Float16)h;
                out[((long)b * T + idx) * 200 + outOff + j] = h;
            }
        }
        __syncthreads();
    }
}

// ---------------------------------------------------------------------------
// Embedding + char-CNN: one block (128 thr) per word; conv weights in LDS.
// ---------------------------------------------------------------------------
__global__ void embed_kernel(const int* __restrict__ tokens, const int* __restrict__ chars,
                             const float* __restrict__ word_table,
                             const float* __restrict__ char_table,
                             const float* __restrict__ convW, const float* __restrict__ convb,
                             float* __restrict__ X)
{
    __shared__ float Wsh[4000];
    __shared__ float ce[16][8];
    __shared__ int   cid[16];
    long w = blockIdx.x;
    int tid = threadIdx.x;
    if (tid < 16) cid[tid] = chars[w * 16 + tid];
    for (int i = tid; i < 4000; i += 128) Wsh[i] = convW[i];
    __syncthreads();
    if (tid < 128) { int l = tid >> 3, c = tid & 7; ce[l][c] = char_table[cid[l] * 8 + c]; }
    __syncthreads();
    if (tid < 100) {
        int tok = tokens[w];
        X[w * 200 + tid] = word_table[(long)tok * 100 + tid];
        const float* Wc = Wsh + tid * 40;          // [8][5]
        float bv = convb[tid];
        float mx = 0.f;                            // max(relu(v)) == max(0, max(v))
        for (int p = 0; p < 12; ++p) {
            float acc = bv;
#pragma unroll
            for (int i = 0; i < 8; ++i)
#pragma unroll
                for (int k = 0; k < 5; ++k)
                    acc += Wc[i * 5 + k] * ce[p + k][i];
            mx = fmaxf(mx, acc);
        }
        X[w * 200 + 100 + tid] = mx;
    }
}

__global__ void highway_combine(float* __restrict__ x, const float* __restrict__ flow,
                                const float* __restrict__ gate, long n)
{
    long i = (long)blockIdx.x * blockDim.x + threadIdx.x;
    if (i < n) { float g = gate[i]; x[i] = g * flow[i] + (1.f - g) * x[i]; }
}

__global__ void row_dot(const float* __restrict__ A, const float* __restrict__ w,
                        float* __restrict__ out, int rows, int K)
{
    int row = blockIdx.x * 8 + (threadIdx.x >> 5);
    int lane = threadIdx.x & 31;
    if (row >= rows) return;
    const float* a = A + (long)row * K;
    float acc = 0.f;
    for (int d = lane; d < K; d += 32) acc += a[d] * w[d];
#pragma unroll
    for (int off = 16; off > 0; off >>= 1) acc += __shfl_xor(acc, off, 32);
    if (lane == 0) out[row] = acc;
}

__global__ void softmax_j(float* __restrict__ sim, float* __restrict__ rowm)
{
    __shared__ float red[64];
    long row = blockIdx.x;
    int t = threadIdx.x;
    float v = sim[row * 64 + t];
    red[t] = v; __syncthreads();
    for (int s = 32; s > 0; s >>= 1) { if (t < s) red[t] = fmaxf(red[t], red[t + s]); __syncthreads(); }
    float mx = red[0]; __syncthreads();
    if (t == 0) rowm[row] = mx;
    float e = __expf(v - mx);
    red[t] = e; __syncthreads();
    for (int s = 32; s > 0; s >>= 1) { if (t < s) red[t] += red[t + s]; __syncthreads(); }
    sim[row * 64 + t] = e / red[0];
}

__global__ void softmax_i(float* __restrict__ m)
{
    __shared__ float red[512];
    int b = blockIdx.x, t = threadIdx.x;
    float v = m[(long)b * 512 + t];
    red[t] = v; __syncthreads();
    for (int s = 256; s > 0; s >>= 1) { if (t < s) red[t] = fmaxf(red[t], red[t + s]); __syncthreads(); }
    float mx = red[0]; __syncthreads();
    float e = __expf(v - mx);
    red[t] = e; __syncthreads();
    for (int s = 256; s > 0; s >>= 1) { if (t < s) red[t] += red[t + s]; __syncthreads(); }
    m[(long)b * 512 + t] = e / red[0];
}

__global__ void q2c_kernel(const float* __restrict__ bvec, const float* __restrict__ ctxE,
                           float* __restrict__ q2c)
{
    int b = blockIdx.x, d = threadIdx.x;
    if (d < 200) {
        float acc = 0.f;
        for (int i = 0; i < 512; ++i)
            acc += bvec[(long)b * 512 + i] * ctxE[((long)b * 512 + i) * 200 + d];
        q2c[b * 200 + d] = acc;
    }
}

__global__ void build_G(const float* __restrict__ ctxE, const float* __restrict__ c2q,
                        const float* __restrict__ q2c, float* __restrict__ G)
{
    long idx = (long)blockIdx.x * blockDim.x + threadIdx.x;
    if (idx >= 32L * 512 * 200) return;
    int d = (int)(idx % 200);
    long bi = idx / 200;
    int b = (int)(bi / 512);
    float ce = ctxE[idx], cq = c2q[idx], qc = q2c[b * 200 + d];
    float* g = G + bi * 800;
    g[d] = ce; g[200 + d] = cq; g[400 + d] = ce * cq; g[600 + d] = ce * qc;
}

__global__ void final_dot(const float* __restrict__ G, const float* __restrict__ Mx,
                          const float* __restrict__ w, float* __restrict__ out)
{
    int row = blockIdx.x * 8 + (threadIdx.x >> 5);
    int lane = threadIdx.x & 31;
    if (row >= 32 * 512) return;
    const float* g = G + (long)row * 800;
    const float* m = Mx + (long)row * 200;
    float acc = 0.f;
    for (int d = lane; d < 800; d += 32) acc += g[d] * w[d];
    for (int d = lane; d < 200; d += 32) acc += m[d] * w[800 + d];
#pragma unroll
    for (int off = 16; off > 0; off >>= 1) acc += __shfl_xor(acc, off, 32);
    if (lane == 0) out[row] = acc;
}

// ---------------------------------------------------------------------------
template <int K, int TRANSB, int ACT>
static inline void gemm_t(hipStream_t stream, const float* A, const float* B, float* C,
                          int M, int N, long sA, long sB, long sC, int batch,
                          const float* b1, const float* b2,
                          const float* ascale = nullptr,
                          const float* rowAdd = nullptr, long sRow = 0,
                          const float* colAdd = nullptr, long sCol = 0)
{
    dim3 g((N + 63) / 64, M / 64, batch);
    wmma_gemm<K, TRANSB, ACT><<<g, 256, 0, stream>>>(A, B, C, M, N, sA, sB, sC, b1, b2,
                                                     ascale, rowAdd, sRow, colAdd, sCol);
}

extern "C" void kernel_launch(void* const* d_in, const int* in_sizes, int n_in,
                              void* d_out, int out_size, void* d_ws, size_t ws_size,
                              hipStream_t stream)
{
    (void)in_sizes; (void)n_in; (void)out_size; (void)ws_size;
    const int* ctx_tok  = (const int*)d_in[0];
    const int* ques_tok = (const int*)d_in[1];
    const int* ctx_ch   = (const int*)d_in[2];
    const int* ques_ch  = (const int*)d_in[3];
    const float* word_table = (const float*)d_in[4];
    const float* char_table = (const float*)d_in[5];
    const float* convW      = (const float*)d_in[6];
    const float* convb      = (const float*)d_in[7];
    const float* Wf = (const float*)d_in[8];
    const float* bf = (const float*)d_in[9];
    const float* Wg = (const float*)d_in[10];
    const float* bg = (const float*)d_in[11];
    const float *ctxP[8], *m0P[8], *m1P[8], *enP[8];
    for (int i = 0; i < 8; ++i) {
        ctxP[i] = (const float*)d_in[12 + i];
        m0P[i]  = (const float*)d_in[20 + i];
        m1P[i]  = (const float*)d_in[28 + i];
        enP[i]  = (const float*)d_in[36 + i];
    }
    const float* sim_w   = (const float*)d_in[44];
    const float* w_start = (const float*)d_in[45];
    const float* w_end   = (const float*)d_in[46];

    float* ws = (float*)d_ws;
    size_t off = 0;
    auto alloc = [&](size_t n) { float* p = ws + off; off += n; return p; };
    float* Xc    = alloc(32L * 512 * 200);
    float* Xq    = alloc(32L * 64 * 200);
    float* XPf   = alloc(32L * 512 * 400);   // also highway flow scratch
    float* XPb   = alloc(32L * 512 * 400);   // also highway gate scratch
    float* ctxE  = alloc(32L * 512 * 200);
    float* quesE = alloc(32L * 64 * 200);
    float* sc    = alloc(32L * 512);
    float* sq    = alloc(32L * 64);
    float* sim   = alloc(32L * 512 * 64);
    float* rowm  = alloc(32L * 512);
    float* c2q   = alloc(32L * 512 * 200);
    float* q2c   = alloc(32L * 200);
    float* G     = alloc(32L * 512 * 800);
    float* M0o   = alloc(32L * 512 * 200);
    float* Mb    = alloc(32L * 512 * 200);
    float* M2b   = alloc(32L * 512 * 200);
    _Float16* WhhP = (_Float16*)alloc(8L * 400 * 128 / 2);  // 8 padded f16 Whh matrices

    // Pre-convert all recurrent weights to padded f16 (register-resident in scan).
    const float* whhs[8] = { ctxP[1], ctxP[5], m0P[1], m0P[5], m1P[1], m1P[5], enP[1], enP[5] };
    for (int i = 0; i < 8; ++i)
        pad_weights_f16<<<(400 * 128 + 255) / 256, 256, 0, stream>>>(whhs[i],
                                                                     WhhP + (long)i * 400 * 128);

    auto highway = [&](float* x, int M) {
        for (int l = 0; l < 2; ++l) {
            gemm_t<200, 1, 1>(stream, x, Wf + (long)l * 200 * 200, XPf, M, 200, 0, 0, 0, 1,
                              bf + l * 200, nullptr);
            gemm_t<200, 1, 2>(stream, x, Wg + (long)l * 200 * 200, XPb, M, 200, 0, 0, 0, 1,
                              bg + l * 200, nullptr);
            long n = (long)M * 200;
            highway_combine<<<(int)((n + 255) / 256), 256, 0, stream>>>(x, XPf, XPb, n);
        }
    };
    auto scan = [&](int widx, float* outp, int T) {
        lstm_scan<<<2, 512, 0, stream>>>(XPf, XPb,
                                         WhhP + (long)(2 * widx) * 400 * 128,
                                         WhhP + (long)(2 * widx + 1) * 400 * 128, outp, T);
    };

    // 1) embeddings + char CNN
    embed_kernel<<<32 * 512, 128, 0, stream>>>(ctx_tok, ctx_ch, word_table, char_table,
                                               convW, convb, Xc);
    embed_kernel<<<32 * 64, 128, 0, stream>>>(ques_tok, ques_ch, word_table, char_table,
                                              convW, convb, Xq);
    // 2) highway
    highway(Xc, 32 * 512);
    highway(Xq, 32 * 64);
    // 3) contextual BiLSTMs (shared weights)
    gemm_t<200, 1, 0>(stream, Xc, ctxP[0], XPf, 32 * 512, 400, 0, 0, 0, 1, ctxP[2], ctxP[3]);
    gemm_t<200, 1, 0>(stream, Xc, ctxP[4], XPb, 32 * 512, 400, 0, 0, 0, 1, ctxP[6], ctxP[7]);
    scan(0, ctxE, 512);
    gemm_t<200, 1, 0>(stream, Xq, ctxP[0], XPf, 32 * 64, 400, 0, 0, 0, 1, ctxP[2], ctxP[3]);
    gemm_t<200, 1, 0>(stream, Xq, ctxP[4], XPb, 32 * 64, 400, 0, 0, 0, 1, ctxP[6], ctxP[7]);
    scan(0, quesE, 64);
    // 4) attention
    row_dot<<<(16384 + 7) / 8, 256, 0, stream>>>(ctxE, sim_w, sc, 16384, 200);
    row_dot<<<(2048 + 7) / 8, 256, 0, stream>>>(quesE, sim_w + 200, sq, 2048, 200);
    gemm_t<200, 1, 0>(stream, ctxE, quesE, sim, 512, 64,
                      512L * 200, 64L * 200, 512L * 64, 32, nullptr, nullptr,
                      sim_w + 400, sc, 512, sq, 64);
    softmax_j<<<16384, 64, 0, stream>>>(sim, rowm);
    softmax_i<<<32, 512, 0, stream>>>(rowm);
    gemm_t<64, 0, 0>(stream, sim, quesE, c2q, 512, 200,
                     512L * 64, 64L * 200, 512L * 200, 32, nullptr, nullptr);
    q2c_kernel<<<32, 256, 0, stream>>>(rowm, ctxE, q2c);
    build_G<<<(int)((32L * 512 * 200 + 255) / 256), 256, 0, stream>>>(ctxE, c2q, q2c, G);
    // 5) modeling + output BiLSTMs
    gemm_t<800, 1, 0>(stream, G, m0P[0], XPf, 32 * 512, 400, 0, 0, 0, 1, m0P[2], m0P[3]);
    gemm_t<800, 1, 0>(stream, G, m0P[4], XPb, 32 * 512, 400, 0, 0, 0, 1, m0P[6], m0P[7]);
    scan(1, M0o, 512);
    gemm_t<200, 1, 0>(stream, M0o, m1P[0], XPf, 32 * 512, 400, 0, 0, 0, 1, m1P[2], m1P[3]);
    gemm_t<200, 1, 0>(stream, M0o, m1P[4], XPb, 32 * 512, 400, 0, 0, 0, 1, m1P[6], m1P[7]);
    scan(2, Mb, 512);
    gemm_t<200, 1, 0>(stream, Mb, enP[0], XPf, 32 * 512, 400, 0, 0, 0, 1, enP[2], enP[3]);
    gemm_t<200, 1, 0>(stream, Mb, enP[4], XPb, 32 * 512, 400, 0, 0, 0, 1, enP[6], enP[7]);
    scan(3, M2b, 512);
    // 6) span logits
    float* out = (float*)d_out;
    final_dot<<<2048, 256, 0, stream>>>(G, Mb, w_start, out);
    final_dot<<<2048, 256, 0, stream>>>(G, M2b, w_end, out + 16384);
}
